// Smartpool_55714315764160
// MI455X (gfx1250) — matrix operand order repaired and checked by reading, not compiled, not run on hardware
//
#include <hip/hip_runtime.h>
#include <hip/hip_bf16.h>

// ---------------- problem constants ----------------
#define BB   16
#define TT   2048
#define CC   512
#define HH   2048
#define NEWT 1024

// GEMM tiling: macro 256(M) x 128(N), 8 waves of 64x64, K-step 32
#define BM 256
#define BN 128
#define BK 32
#define ASTR 40              // LDS A row stride (ushorts): 32 data + 4 pad-pairs
#define BSTR 20              // LDS B row stride (uints):   16 data + 4 pad
#define ABUF (BM * ASTR * 2) // 20480 B per A buffer
#define BBUF (BN * BSTR * 4) // 10240 B per B buffer
#define SMEMB (2 * (ABUF + BBUF)) // 61440 B (double buffered); red[256*32]f aliases
#define NPART 16             // = HH / BN : partials per row from GEMM2

typedef __attribute__((ext_vector_type(16))) __bf16   v16bf;
typedef __attribute__((ext_vector_type(8)))  float    v8f;
typedef __attribute__((ext_vector_type(4)))  unsigned u32x4v;
typedef __attribute__((ext_vector_type(8)))  int      i32x8v;
typedef __attribute__((ext_vector_type(4)))  int      i32x4v;

#if defined(__has_builtin)
#if __has_builtin(__builtin_amdgcn_tensor_load_to_lds)
#define USE_TDM 1
#endif
#endif
#ifndef USE_TDM
#define USE_TDM 0
#endif

__device__ __forceinline__ float gelu_exact(float x) {
    return 0.5f * x * (1.0f + erff(x * 0.70710678118654752440f));
}

#if USE_TDM
// 2D tile load via Tensor Data Mover. Descriptor per CDNA5 ISA sec 8.3/8.4.
// Rows of 64B get 16B LDS padding (pad_interval=3 -> 16 dwords, pad_amount=3 -> 4 dwords).
__device__ __forceinline__ void tdm_load_2d(unsigned lds_addr, const void* gptr,
                                            unsigned data_size /*1=2B,2=4B*/,
                                            unsigned tile_d0, unsigned tile_d1,
                                            unsigned tensor_d0, unsigned tensor_d1,
                                            unsigned long long stride0) {
    unsigned long long ga = (unsigned long long)gptr;
    u32x4v g0 = {};
    g0[0] = 1u;                                            // count=1, user mode
    g0[1] = lds_addr;                                      // LDS byte address
    g0[2] = (unsigned)ga;                                  // global_addr[31:0]
    g0[3] = (unsigned)((ga >> 32) & 0x1FFFFFFu) | (2u << 30); // global_addr[56:32], type=2
    i32x8v g1 = {};
    g1[0] = (int)((data_size << 16) | (1u << 20) | (3u << 22) | (3u << 25)); // pad en/interval/amount
    g1[1] = (int)((tensor_d0 & 0xFFFFu) << 16);            // tensor_dim0[15:0] @bit48
    g1[2] = (int)((tensor_d0 >> 16) | ((tensor_d1 & 0xFFFFu) << 16));
    g1[3] = (int)((tensor_d1 >> 16) | (tile_d0 << 16));    // tile_dim0 @bit112
    g1[4] = (int)(tile_d1 & 0xFFFFu);                      // tile_dim1; tile_dim2=0
    g1[5] = (int)(unsigned)(stride0 & 0xFFFFFFFFu);        // tensor_dim0_stride[31:0]
    g1[6] = (int)(unsigned)((stride0 >> 32) & 0xFFFFu);    // stride[47:32]; dim1_stride=0 (2D)
    g1[7] = 0;
    i32x4v gz4 = {};   // D# groups 2/3: unused for 2D tile
    i32x8v gz8 = {};   // extra descriptor group (6-arg builtin form): unused
    __builtin_amdgcn_tensor_load_to_lds(g0, g1, gz4, gz4, gz8, 0);
}
#endif

// ---------------- f32 -> bf16 convert ----------------
__global__ void k_cvt_bf16(const float* __restrict__ in, __bf16* __restrict__ out, int n) {
    int i = blockIdx.x * blockDim.x + threadIdx.x;
    if (i < n) out[i] = (__bf16)in[i];
}

// ---------------- W [K,N] f32 -> Wp [N][K/2] uint (bf16 pair packed along K) ----------------
__global__ void k_packW(const float* __restrict__ W, unsigned* __restrict__ Wp, int K, int N) {
    int id = blockIdx.x * blockDim.x + threadIdx.x;
    int Kd2 = K >> 1;
    if (id >= N * Kd2) return;
    int n = id / Kd2, kp = id - n * Kd2;
    union { __bf16 h[2]; unsigned u; } pk;
    pk.h[0] = (__bf16)W[(size_t)(2 * kp    ) * N + n];
    pk.h[1] = (__bf16)W[(size_t)(2 * kp + 1) * N + n];
    Wp[(size_t)n * Kd2 + kp] = pk.u;
}

// ---------------- bf16 WMMA GEMM ----------------
// A [M,K] bf16 row-major; Bp [N][K/2] packed. grid=(M/256, N/128), 256 thr (8 waves, 64x64 each).
// EPI=0: Hout = gelu(A@B + bias) bf16.  EPI=1: partials[m][ny] = sum_n gelu(..)*w3[n].
template<int EPI>
__global__ __launch_bounds__(256) void k_gemm_bf16(
    const unsigned short* __restrict__ A, const unsigned* __restrict__ Bp,
    const float* __restrict__ bias, const float* __restrict__ w3,
    __bf16* __restrict__ Hout, float* __restrict__ partials,
    int K, int N)
{
    __shared__ __align__(16) char smem[SMEMB];

    const int tid  = threadIdx.x;
    const int lane = tid & 31, wave = tid >> 5;
    const int wm = wave & 3, wn = wave >> 2;        // wave grid 4(M) x 2(N)
    const int half = lane >> 4, lr = lane & 15;
    const int m0 = blockIdx.x * BM, n0 = blockIdx.y * BN;
    const int Kd2 = K >> 1;
    const int nk = K / BK;

    v8f acc[4][4] = {};
    union Frag { v16bf v; u32x4v u[2]; };

#if USE_TDM
    const unsigned lds0 = (unsigned)(size_t)(void*)smem;   // LDS byte offset (low 32 bits)
    const unsigned ldsA[2] = { lds0,             lds0 + ABUF };
    const unsigned ldsB[2] = { lds0 + 2 * ABUF,  lds0 + 2 * ABUF + BBUF };
    const int Mrows = gridDim.x * BM;

    if (wave == 0) {  // prologue: stage tile 0, wait, publish
        tdm_load_2d(ldsA[0], (const char*)A + ((size_t)m0 * K) * 2, 1, BK, BM,
                    (unsigned)K, (unsigned)Mrows, (unsigned long long)K);
        tdm_load_2d(ldsB[0], (const char*)Bp + ((size_t)n0 * Kd2) * 4, 2, BK / 2, BN,
                    (unsigned)Kd2, (unsigned)N, (unsigned long long)Kd2);
        __builtin_amdgcn_s_wait_tensorcnt(0);
    }
    __syncthreads();

    for (int ks = 0; ks < nk; ++ks) {
        const int p = ks & 1;
        if (wave == 0 && ks + 1 < nk) {  // async-stage next tile into other buffer
            const int k0n = (ks + 1) * BK;
            tdm_load_2d(ldsA[1 - p], (const char*)A + ((size_t)m0 * K + k0n) * 2, 1, BK, BM,
                        (unsigned)K, (unsigned)Mrows, (unsigned long long)K);
            tdm_load_2d(ldsB[1 - p], (const char*)Bp + ((size_t)n0 * Kd2 + (k0n >> 1)) * 4, 2,
                        BK / 2, BN, (unsigned)Kd2, (unsigned)N, (unsigned long long)Kd2);
        }
        const unsigned short* As = (const unsigned short*)(smem + p * ABUF);
        const unsigned*       Bs = (const unsigned*)(smem + 2 * ABUF + p * BBUF);

        Frag a[4], b[4];
        #pragma unroll
        for (int mt = 0; mt < 4; ++mt) {
            int r = (wm * 64 + mt * 16 + lr) * ASTR + half * 8;
            a[mt].u[0] = *(const u32x4v*)&As[r];
            a[mt].u[1] = *(const u32x4v*)&As[r + 16];
        }
        #pragma unroll
        for (int nt = 0; nt < 4; ++nt) {
            int r = (wn * 64 + nt * 16 + lr) * BSTR + half * 4;
            b[nt].u[0] = *(const u32x4v*)&Bs[r];
            b[nt].u[1] = *(const u32x4v*)&Bs[r + 8];
        }
        #pragma unroll
        for (int mt = 0; mt < 4; ++mt)
            #pragma unroll
            for (int nt = 0; nt < 4; ++nt)
                acc[mt][nt] = __builtin_amdgcn_wmma_f32_16x16x32_bf16(
                    false, a[mt].v, false, b[nt].v, (short)0, acc[mt][nt], false, false);

        if (wave == 0) __builtin_amdgcn_s_wait_tensorcnt(0);
        __syncthreads();
    }
#else
    // manual staging fallback (single buffer, two barriers per k-step)
    unsigned short* As = (unsigned short*)smem;
    unsigned*       Bs = (unsigned*)(smem + 2 * ABUF);
    const u32x4v* A4 = (const u32x4v*)A;
    const u32x4v* B4 = (const u32x4v*)Bp;
    for (int ks = 0; ks < nk; ++ks) {
        const int k0 = ks * BK;
        #pragma unroll
        for (int s = 0; s < 4; ++s) {
            int c = tid + s * 256;           // 1024 chunks: 256 rows x 4 x 16B
            int row = c >> 2, off = c & 3;
            *(u32x4v*)&As[row * ASTR + off * 8] = A4[((size_t)(m0 + row) * K + k0) / 8 + off];
        }
        #pragma unroll
        for (int s = 0; s < 2; ++s) {
            int c = tid + s * 256;           // 512 chunks: 128 rows x 4 x 16B
            int n = c >> 2, off = c & 3;
            *(u32x4v*)&Bs[n * BSTR + off * 4] = B4[((size_t)(n0 + n) * Kd2 + (k0 >> 1)) / 4 + off];
        }
        __syncthreads();
        Frag a[4], b[4];
        #pragma unroll
        for (int mt = 0; mt < 4; ++mt) {
            int r = (wm * 64 + mt * 16 + lr) * ASTR + half * 8;
            a[mt].u[0] = *(const u32x4v*)&As[r];
            a[mt].u[1] = *(const u32x4v*)&As[r + 16];
        }
        #pragma unroll
        for (int nt = 0; nt < 4; ++nt) {
            int r = (wn * 64 + nt * 16 + lr) * BSTR + half * 4;
            b[nt].u[0] = *(const u32x4v*)&Bs[r];
            b[nt].u[1] = *(const u32x4v*)&Bs[r + 8];
        }
        #pragma unroll
        for (int mt = 0; mt < 4; ++mt)
            #pragma unroll
            for (int nt = 0; nt < 4; ++nt)
                acc[mt][nt] = __builtin_amdgcn_wmma_f32_16x16x32_bf16(
                    false, a[mt].v, false, b[nt].v, (short)0, acc[mt][nt], false, false);
        __syncthreads();
    }
#endif

    if (EPI == 0) {
        #pragma unroll
        for (int nt = 0; nt < 4; ++nt) {
            int n = n0 + wn * 64 + nt * 16 + lr;
            float bv = bias[n];
            #pragma unroll
            for (int mt = 0; mt < 4; ++mt) {
                int mbase = m0 + wm * 64 + mt * 16 + half * 8;
                #pragma unroll
                for (int i = 0; i < 8; ++i) {
                    float v = gelu_exact(acc[mt][nt][i] + bv);
                    Hout[(size_t)(mbase + i) * N + n] = (__bf16)v;
                }
            }
        }
    } else {
        float* red = (float*)smem;  // [256][32], aliases staging buffers (after final barrier)
        float bv[4], wv[4];
        #pragma unroll
        for (int nt = 0; nt < 4; ++nt) {
            int n = n0 + wn * 64 + nt * 16 + lr;
            bv[nt] = bias[n];
            wv[nt] = w3[n];
        }
        #pragma unroll
        for (int mt = 0; mt < 4; ++mt) {
            #pragma unroll
            for (int i = 0; i < 8; ++i) {
                float s = 0.f;
                #pragma unroll
                for (int nt = 0; nt < 4; ++nt)
                    s += gelu_exact(acc[mt][nt][i] + bv[nt]) * wv[nt];
                int mloc = wm * 64 + mt * 16 + half * 8 + i;
                red[mloc * 32 + (wn * 16 + lr)] = s;  // 256x32 cells, each written once
            }
        }
        __syncthreads();
        {
            float s = 0.f;
            #pragma unroll
            for (int c = 0; c < 32; ++c) s += red[tid * 32 + c];
            partials[(size_t)(m0 + tid) * gridDim.y + blockIdx.y] = s;
        }
    }
}

// ---------------- per-batch-row: logits -> sigmoid -> normalize to T/2 -> cumsum ----------------
__global__ __launch_bounds__(256) void k_newlens(const float* __restrict__ partials,
                                                 const float* __restrict__ b3,
                                                 float* __restrict__ cs) {
    __shared__ float sc[256];
    const int b = blockIdx.x, tid = threadIdx.x;
    const float bias = b3[0];
    float vals[8];
    float ls = 0.f;
    #pragma unroll
    for (int k = 0; k < 8; ++k) {
        int t = tid * 8 + k;
        size_t row = (size_t)b * TT + t;
        float s = 0.f;
        #pragma unroll
        for (int p = 0; p < NPART; ++p) s += partials[row * NPART + p];
        float x = 1.f / (1.f + expf(-(s + bias)));
        vals[k] = x; ls += x;
    }
    sc[tid] = ls;
    __syncthreads();
    #pragma unroll
    for (int off = 1; off < 256; off <<= 1) {
        float add = (tid >= off) ? sc[tid - off] : 0.f;
        __syncthreads();
        sc[tid] += add;
        __syncthreads();
    }
    float total = sc[255];
    float scale = (float)(TT / 2) / total;
    float run = sc[tid] - ls;
    #pragma unroll
    for (int k = 0; k < 8; ++k) {
        run += vals[k];
        cs[(size_t)b * TT + tid * 8 + k] = run * scale;
    }
}

// ---------------- warp: banded resample, out[b,j,:] = sum_i w(i,j) * X[b,i,:] ----------------
__global__ __launch_bounds__(256) void k_warp(const float* __restrict__ X,
                                              const float* __restrict__ cs,
                                              float* __restrict__ out) {
    const int b = blockIdx.y, j = blockIdx.x;
    const float* csr = cs + (size_t)b * TT;
    const float fj = (float)j, fj1 = fj + 1.f;
    int lo = 0, hi = TT;
    while (lo < hi) { int mid = (lo + hi) >> 1; if (csr[mid] > fj) hi = mid; else lo = mid + 1; }
    int i = lo;
    const int c0 = threadIdx.x * 2;
    const float* Xb = X + (size_t)b * TT * CC;
    float a0 = 0.f, a1 = 0.f;
    float cprev = (i == 0) ? 0.f : csr[i - 1];
    while (i < TT && cprev < fj1) {
        float ci = csr[i];
        float wa = fminf(fmaxf(ci    - fj, 0.f), 1.f);
        float wb = fminf(fmaxf(cprev - fj, 0.f), 1.f);
        float w  = wa - wb;
        const float2 xv = *(const float2*)(Xb + (size_t)i * CC + c0);
        a0 = fmaf(w, xv.x, a0);
        a1 = fmaf(w, xv.y, a1);
        cprev = ci; ++i;
    }
    float2 r; r.x = a0; r.y = a1;
    *(float2*)(out + ((size_t)b * NEWT + j) * CC + c0) = r;
}

// ---------------- launcher ----------------
extern "C" void kernel_launch(void* const* d_in, const int* in_sizes, int n_in,
                              void* d_out, int out_size, void* d_ws, size_t ws_size,
                              hipStream_t stream) {
    const float* feat = (const float*)d_in[0];
    const float* W1   = (const float*)d_in[1];
    const float* b1   = (const float*)d_in[2];
    const float* W2   = (const float*)d_in[3];
    const float* b2   = (const float*)d_in[4];
    const float* W3   = (const float*)d_in[5];
    const float* b3   = (const float*)d_in[6];

    char* ws = (char*)d_ws;
    const size_t szFeatA = (size_t)BB * TT * CC * 2;
    const size_t szW1p   = (size_t)HH * (CC / 2) * 4;
    const size_t szW2p   = (size_t)HH * (HH / 2) * 4;
    const size_t szH1    = (size_t)BB * TT * HH * 2;
    const size_t szPart  = (size_t)BB * TT * NPART * 4;
    __bf16*   featA = (__bf16*)(ws);
    unsigned* W1p   = (unsigned*)(ws + szFeatA);
    unsigned* W2p   = (unsigned*)(ws + szFeatA + szW1p);
    __bf16*   h1    = (__bf16*)(ws + szFeatA + szW1p + szW2p);
    float*    part  = (float*)(ws + szFeatA + szW1p + szW2p + szH1);
    float*    cs    = (float*)(ws + szFeatA + szW1p + szW2p + szH1 + szPart);

    {
        int n = BB * TT * CC;
        k_cvt_bf16<<<(n + 255) / 256, 256, 0, stream>>>(feat, featA, n);
        int n1 = HH * (CC / 2);
        k_packW<<<(n1 + 255) / 256, 256, 0, stream>>>(W1, W1p, CC, HH);
        int n2 = HH * (HH / 2);
        k_packW<<<(n2 + 255) / 256, 256, 0, stream>>>(W2, W2p, HH, HH);
    }
    {
        dim3 grid((BB * TT) / BM, HH / BN);
        k_gemm_bf16<0><<<grid, 256, 0, stream>>>((const unsigned short*)featA, W1p,
                                                 b1, nullptr, h1, nullptr, CC, HH);
    }
    {
        dim3 grid((BB * TT) / BM, HH / BN);
        k_gemm_bf16<1><<<grid, 256, 0, stream>>>((const unsigned short*)h1, W2p,
                                                 b2, W3, nullptr, part, HH, HH);
    }
    k_newlens<<<BB, 256, 0, stream>>>(part, b3, cs);
    {
        dim3 grid(NEWT, BB);
        k_warp<<<grid, 256, 0, stream>>>(feat, cs, (float*)d_out);
    }
    (void)in_sizes; (void)n_in; (void)out_size; (void)ws_size;
}
